// Attention_27633819582561
// MI455X (gfx1250) — matrix operand order repaired and checked by reading, not compiled, not run on hardware
//
#include <hip/hip_runtime.h>
#include <hip/hip_bf16.h>
#include <stddef.h>

// ---------------------------------------------------------------------------
// SAM decomposed rel-pos attention for MI455X (gfx1250, wave32, WMMA).
// bf16 matmuls via v_wmma_f32_16x16x32_bf16, fp32 softmax/accum,
// flash-attention fusion so the 510MB attn matrix never touches HBM.
// ---------------------------------------------------------------------------

typedef __bf16 bf16_t;
typedef bf16_t v8bf  __attribute__((ext_vector_type(8)));
typedef bf16_t v16bf __attribute__((ext_vector_type(16)));
typedef float  v8f   __attribute__((ext_vector_type(8)));

#define BATCH   2
#define HDIM    48
#define WDIM    48
#define NSEQ    (HDIM * WDIM)      // 2304
#define CDIM    768
#define HD      64
#define NH      12
#define BH      (BATCH * NH)       // 24
#define MROWS   (BATCH * NSEQ)     // 4608

__device__ __forceinline__ v16bf cat16(v8bf lo, v8bf hi) {
  return __builtin_shufflevector(lo, hi, 0,1,2,3,4,5,6,7,8,9,10,11,12,13,14,15);
}

// A-matrix fragment (16x32 bf16), source row-major [row, k], leading dim ld.
// Lanes 0-15 hold K {0..7, 16..23}; lanes 16-31 hold K {8..15, 24..31}.
__device__ __forceinline__ v16bf load_a(const bf16_t* __restrict__ base, int ld,
                                        int row, int k0, int lane) {
  const bf16_t* p = base + (size_t)row * ld + k0 + ((lane & 16) ? 8 : 0);
  v8bf lo = *(const v8bf*)p;
  v8bf hi = *(const v8bf*)(p + 16);
  return cat16(lo, hi);
}

// B-matrix fragment (32x16 bf16). Source holds B^T row-major: [n, k], ld.
// Lanes 0-15 hold K {0..7, 16..23}; lanes 16-31 hold K {8..15, 24..31} (col = lane&15).
__device__ __forceinline__ v16bf load_b(const bf16_t* __restrict__ base, int ld,
                                        int col, int k0, int lane) {
  const bf16_t* p = base + (size_t)col * ld + k0 + ((lane & 16) ? 8 : 0);
  v8bf lo = *(const v8bf*)p;
  v8bf hi = *(const v8bf*)(p + 16);
  return cat16(lo, hi);
}

__device__ __forceinline__ v8f wmma_bf16(v16bf a, v16bf b, v8f c) {
  return __builtin_amdgcn_wmma_f32_16x16x32_bf16(
      /*neg_a=*/false, a, /*neg_b=*/false, b,
      /*c_mod=*/(short)0, c, /*reuse_a=*/false, /*reuse_b=*/false);
}

// ---------------------------------------------------------------------------
// fp32 -> bf16 conversion
// ---------------------------------------------------------------------------
__global__ void cvt_f32_bf16_kernel(const float* __restrict__ in,
                                    bf16_t* __restrict__ out, int n) {
  int i = blockIdx.x * 256 + threadIdx.x;
  if (i < n) out[i] = (bf16_t)in[i];
}

// ---------------------------------------------------------------------------
// WMMA GEMM: out[m,n] = sum_k A[m,k] * W[n,k] + bias[n]
// block = 256 threads (8 waves), block tile 128x64, wave tile 32x32.
// MODE 0: QKV epilogue scatters into Q / K (row-major [N,64]) and V^T ([64,N])
// MODE 1: proj epilogue writes fp32 d_out.
// ---------------------------------------------------------------------------
template <int MODE>
__global__ __launch_bounds__(256) void gemm_bf16_kernel(
    const bf16_t* __restrict__ A, const bf16_t* __restrict__ W,
    const float* __restrict__ bias,
    bf16_t* __restrict__ qb, bf16_t* __restrict__ kb, bf16_t* __restrict__ vtb,
    float* __restrict__ outf, int Nd, int K) {
  const int lane = threadIdx.x & 31;
  const int wave = threadIdx.x >> 5;
  const int wm = wave >> 1, wn = wave & 1;
  const int mbase = blockIdx.y * 128 + wm * 32;
  const int nbase = blockIdx.x * 64 + wn * 32;
  const int rlo = lane & 15;
  const int rhi = (lane & 16) ? 8 : 0;

  v8f acc00 = {}, acc01 = {}, acc10 = {}, acc11 = {};

  for (int k0 = 0; k0 < K; k0 += 32) {
    if (k0 + 32 < K) {  // stream next K slab into L2 (global_prefetch_b8)
      __builtin_prefetch(A + (size_t)(mbase + rlo) * K + k0 + 32, 0, 1);
      __builtin_prefetch(W + (size_t)(nbase + rlo) * K + k0 + 32, 0, 1);
    }
    v16bf a0 = load_a(A, K, mbase + rlo, k0, lane);
    v16bf a1 = load_a(A, K, mbase + 16 + rlo, k0, lane);
    v16bf b0 = load_b(W, K, nbase + rlo, k0, lane);
    v16bf b1 = load_b(W, K, nbase + 16 + rlo, k0, lane);
    acc00 = wmma_bf16(a0, b0, acc00);
    acc01 = wmma_bf16(a0, b1, acc01);
    acc10 = wmma_bf16(a1, b0, acc10);
    acc11 = wmma_bf16(a1, b1, acc11);
  }

  v8f accs[2][2] = {{acc00, acc01}, {acc10, acc11}};
#pragma unroll
  for (int tm = 0; tm < 2; ++tm) {
#pragma unroll
    for (int tn = 0; tn < 2; ++tn) {
      const int n = nbase + tn * 16 + rlo;
      const float bv = bias[n];
#pragma unroll
      for (int j = 0; j < 8; ++j) {
        const int m = mbase + tm * 16 + j + rhi;
        const float v = accs[tm][tn][j] + bv;
        if (MODE == 0) {
          const int batch = m / NSEQ;
          const int rn    = m - batch * NSEQ;
          const int part  = n / CDIM;         // 0=q 1=k 2=v
          const int c     = n - part * CDIM;
          const int head  = c >> 6;
          const int d     = c & 63;
          const size_t bh = (size_t)batch * NH + head;
          const bf16_t bvv = (bf16_t)v;
          if (part == 0)      qb[(bh * NSEQ + rn) * HD + d] = bvv;
          else if (part == 1) kb[(bh * NSEQ + rn) * HD + d] = bvv;
          else                vtb[(bh * HD + d) * NSEQ + rn] = bvv;
        } else {
          outf[(size_t)m * Nd + n] = v;
        }
      }
    }
  }
}

// ---------------------------------------------------------------------------
// Decomposed rel-pos bias tables:
//   relH[bh,q,kh] = sum_d Q[bh,q,d] * rel_pos_h[qh-kh+47, d]   (q = qh*48+qw)
//   relW[bh,q,kw] = sum_d Q[bh,q,d] * rel_pos_w[qw-kw+47, d]
// one thread per output element (t in [0,96): 0-47 -> H table, 48-95 -> W).
// ---------------------------------------------------------------------------
__global__ __launch_bounds__(256) void relpos_kernel(
    const bf16_t* __restrict__ Q, const float* __restrict__ rel_pos_h,
    const float* __restrict__ rel_pos_w, float* __restrict__ relH,
    float* __restrict__ relW) {
  const int idx = blockIdx.x * 256 + threadIdx.x;   // [BH][NSEQ][96]
  const int t  = idx % 96;
  const int q  = (idx / 96) % NSEQ;
  const int bh = idx / (96 * NSEQ);
  const int qh = q / WDIM, qw = q - qh * WDIM;

  const bf16_t* qp = Q + ((size_t)bh * NSEQ + q) * HD;
  const bool isH = (t < 48);
  const int kidx = isH ? t : (t - 48);
  const int qc   = isH ? qh : qw;
  const float* rp = (isH ? rel_pos_h : rel_pos_w) + (size_t)(qc - kidx + HDIM - 1) * HD;

  float s = 0.f;
#pragma unroll 8
  for (int d = 0; d < HD; ++d) s += (float)qp[d] * rp[d];

  float* dst = isH ? relH : relW;
  dst[((size_t)bh * NSEQ + q) * 48 + kidx] = s;
}

// ---------------------------------------------------------------------------
// Flash attention with decomposed rel-pos bias.
// grid = (NSEQ/64, BH), block = 128 (4 waves), 16 query rows per wave.
// Key loop in chunks of 32: 4 WMMAs for S, online softmax, P through LDS,
// 4 WMMAs for O += P@V.
// ---------------------------------------------------------------------------
__global__ __launch_bounds__(128) void flash_attn_kernel(
    const bf16_t* __restrict__ Q, const bf16_t* __restrict__ K,
    const bf16_t* __restrict__ Vt, const float* __restrict__ relH,
    const float* __restrict__ relW, bf16_t* __restrict__ Obuf) {
  __shared__ bf16_t psh[4][16][48];  // 48-elem (96B) row stride: 16B aligned, spread banks

  const int lane = threadIdx.x & 31;
  const int wave = threadIdx.x >> 5;
  const int bh = blockIdx.y;
  const int batch = bh / NH, head = bh - batch * NH;
  const int q0 = blockIdx.x * 64 + wave * 16;
  const int rlo = lane & 15;
  const int rhi = (lane & 16) ? 8 : 0;

  const bf16_t* Qb = Q  + (size_t)bh * NSEQ * HD;
  const bf16_t* Kb = K  + (size_t)bh * NSEQ * HD;
  const bf16_t* Vb = Vt + (size_t)bh * HD * NSEQ;
  const float* rHb = relH + (size_t)bh * NSEQ * 48;
  const float* rWb = relW + (size_t)bh * NSEQ * 48;

  // Q tile (16 rows x 64) as two A fragments, kept in registers all loop long.
  const v16bf qf0 = load_a(Qb, HD, q0 + rlo, 0, lane);
  const v16bf qf1 = load_a(Qb, HD, q0 + rlo, 32, lane);

  v8f o0 = {}, o1 = {}, o2 = {}, o3 = {};
  float mrow[8], lrow[8];
#pragma unroll
  for (int j = 0; j < 8; ++j) { mrow[j] = -1e30f; lrow[j] = 0.f; }

  const float scale = 0.125f;  // 64^-0.5

  for (int kc = 0; kc < NSEQ; kc += 32) {
    // ---- S = scale * Q @ K^T  (16 x 32) ----
    v8f s0 = {}, s1 = {};
    {
      v16bf kf = load_b(Kb, HD, kc + rlo, 0, lane);
      s0 = wmma_bf16(qf0, kf, s0);
      kf = load_b(Kb, HD, kc + rlo, 32, lane);
      s0 = wmma_bf16(qf1, kf, s0);
      kf = load_b(Kb, HD, kc + 16 + rlo, 0, lane);
      s1 = wmma_bf16(qf0, kf, s1);
      kf = load_b(Kb, HD, kc + 16 + rlo, 32, lane);
      s1 = wmma_bf16(qf1, kf, s1);
    }

    const int kcol0 = kc + rlo, kcol1 = kc + 16 + rlo;
    const int kh0 = kcol0 / WDIM, kw0 = kcol0 - kh0 * WDIM;
    const int kh1 = kcol1 / WDIM, kw1 = kcol1 - kh1 * WDIM;

    float p0[8], p1[8];
#pragma unroll
    for (int j = 0; j < 8; ++j) {
      const int qrow = q0 + j + rhi;
      const float b0 = rHb[(size_t)qrow * 48 + kh0] + rWb[(size_t)qrow * 48 + kw0];
      const float b1 = rHb[(size_t)qrow * 48 + kh1] + rWb[(size_t)qrow * 48 + kw1];
      p0[j] = s0[j] * scale + b0;
      p1[j] = s1[j] * scale + b1;

      // row max across the 16 lanes holding this row (xor butterfly, half-wave)
      float mx = fmaxf(p0[j], p1[j]);
#pragma unroll
      for (int msk = 1; msk < 16; msk <<= 1) mx = fmaxf(mx, __shfl_xor(mx, msk, 32));
      const float mnew = fmaxf(mrow[j], mx);
      const float alpha = __expf(mrow[j] - mnew);
      p0[j] = __expf(p0[j] - mnew);
      p1[j] = __expf(p1[j] - mnew);
      float rs = p0[j] + p1[j];
#pragma unroll
      for (int msk = 1; msk < 16; msk <<= 1) rs += __shfl_xor(rs, msk, 32);
      lrow[j] = lrow[j] * alpha + rs;
      mrow[j] = mnew;
      o0[j] *= alpha; o1[j] *= alpha; o2[j] *= alpha; o3[j] *= alpha;
    }

    // ---- stage P (16x32) into LDS row-major, re-read in A-fragment order ----
#pragma unroll
    for (int j = 0; j < 8; ++j) {
      psh[wave][j + rhi][rlo]      = (bf16_t)p0[j];
      psh[wave][j + rhi][16 + rlo] = (bf16_t)p1[j];
    }
    const bf16_t* pp = &psh[wave][rlo][(lane & 16) ? 8 : 0];
    const v16bf pf = cat16(*(const v8bf*)pp, *(const v8bf*)(pp + 16));

    // ---- O += P @ V  (V^T rows give contiguous-K B fragments) ----
    o0 = wmma_bf16(pf, load_b(Vb, NSEQ,  0 + rlo, kc, lane), o0);
    o1 = wmma_bf16(pf, load_b(Vb, NSEQ, 16 + rlo, kc, lane), o1);
    o2 = wmma_bf16(pf, load_b(Vb, NSEQ, 32 + rlo, kc, lane), o2);
    o3 = wmma_bf16(pf, load_b(Vb, NSEQ, 48 + rlo, kc, lane), o3);
  }

  // ---- normalize and write O as bf16 into [B, N, C] for the proj GEMM ----
  v8f os[4] = {o0, o1, o2, o3};
#pragma unroll
  for (int t = 0; t < 4; ++t) {
#pragma unroll
    for (int j = 0; j < 8; ++j) {
      const int qrow = q0 + j + rhi;
      const size_t row = (size_t)batch * NSEQ + qrow;
      const int col = head * HD + t * 16 + rlo;
      Obuf[row * CDIM + col] = (bf16_t)(os[t][j] / lrow[j]);
    }
  }
}

// ---------------------------------------------------------------------------
// host launcher
// ---------------------------------------------------------------------------
extern "C" void kernel_launch(void* const* d_in, const int* in_sizes, int n_in,
                              void* d_out, int out_size, void* d_ws, size_t ws_size,
                              hipStream_t stream) {
  (void)in_sizes; (void)n_in; (void)out_size; (void)ws_size;
  const float* x         = (const float*)d_in[0];  // [B,N,C]
  const float* qkv_w     = (const float*)d_in[1];  // [3C,C]
  const float* qkv_b     = (const float*)d_in[2];  // [3C]
  const float* proj_w    = (const float*)d_in[3];  // [C,C]
  const float* proj_b    = (const float*)d_in[4];  // [C]
  const float* rel_pos_h = (const float*)d_in[5];  // [95,64]
  const float* rel_pos_w = (const float*)d_in[6];  // [95,64]

  char* ws = (char*)d_ws;
  size_t off = 0;
  auto carve = [&](size_t bytes) -> char* {
    char* p = ws + off;
    off = (off + bytes + 255) & ~(size_t)255;
    return p;
  };
  bf16_t* xb    = (bf16_t*)carve((size_t)MROWS * CDIM * 2);        // x bf16
  bf16_t* wqkv  = (bf16_t*)carve((size_t)3 * CDIM * CDIM * 2);     // qkv_w bf16
  bf16_t* wproj = (bf16_t*)carve((size_t)CDIM * CDIM * 2);         // proj_w bf16
  bf16_t* Qb    = (bf16_t*)carve((size_t)BH * NSEQ * HD * 2);
  bf16_t* Kb    = (bf16_t*)carve((size_t)BH * NSEQ * HD * 2);
  bf16_t* Vtb   = (bf16_t*)carve((size_t)BH * HD * NSEQ * 2);
  float*  relH  = (float*)carve((size_t)BH * NSEQ * 48 * 4);
  float*  relW  = (float*)carve((size_t)BH * NSEQ * 48 * 4);
  bf16_t* Ob    = (bf16_t*)carve((size_t)MROWS * CDIM * 2);

  // 1. fp32 -> bf16 conversions
  {
    const int nx = MROWS * CDIM;          // 3538944
    const int nw = 3 * CDIM * CDIM;       // 1769472
    const int np = CDIM * CDIM;           // 589824
    cvt_f32_bf16_kernel<<<nx / 256, 256, 0, stream>>>(x, xb, nx);
    cvt_f32_bf16_kernel<<<nw / 256, 256, 0, stream>>>(qkv_w, wqkv, nw);
    cvt_f32_bf16_kernel<<<np / 256, 256, 0, stream>>>(proj_w, wproj, np);
  }

  // 2. QKV GEMM with head-reorg scatter epilogue
  gemm_bf16_kernel<0><<<dim3(3 * CDIM / 64, MROWS / 128), 256, 0, stream>>>(
      xb, wqkv, qkv_b, Qb, Kb, Vtb, nullptr, 3 * CDIM, CDIM);

  // 3. rel-pos bias tables
  {
    const int total = BH * NSEQ * 96;  // 5308416, divisible by 256
    relpos_kernel<<<total / 256, 256, 0, stream>>>(Qb, rel_pos_h, rel_pos_w,
                                                   relH, relW);
  }

  // 4. flash attention
  flash_attn_kernel<<<dim3(NSEQ / 64, BH), 128, 0, stream>>>(Qb, Kb, Vtb, relH,
                                                             relW, Ob);

  // 5. output projection -> fp32 d_out
  gemm_bf16_kernel<1><<<dim3(CDIM / 64, MROWS / 128), 256, 0, stream>>>(
      Ob, wproj, proj_b, nullptr, nullptr, nullptr, (float*)d_out, CDIM, CDIM);
}
